// MultiHeadAttention_40991167873488
// MI455X (gfx1250) — matrix operand order repaired and checked by reading, not compiled, not run on hardware
//
#include <hip/hip_runtime.h>

// Problem constants
#define DKC 256
#define NHEAD 8
#define DHEAD 32
#define NBATCH 8
#define NSEQ 128
#define NAGENT 64
#define MTOT (NBATCH*NSEQ*NAGENT)   // 65536 rows for the big GEMMs
#define OUT_ELEMS 16777216          // B*S*N*DK
#define WELEMS 33554432             // B*H*S*N*N

typedef __attribute__((ext_vector_type(16))) __bf16       v16bf;
typedef __attribute__((ext_vector_type(4)))  __bf16       v4bf;
typedef __attribute__((ext_vector_type(8)))  float        v8f;
typedef __attribute__((ext_vector_type(4)))  float        v4f;
typedef __attribute__((ext_vector_type(4)))  unsigned int v4u;

union Frag {
  v16bf b;
  v4bf  b4[4];
  v4u   q[2];
};

__device__ __forceinline__ v8f vzero8() {
  v8f z = {0.f,0.f,0.f,0.f,0.f,0.f,0.f,0.f};
  return z;
}

__device__ __forceinline__ v8f wmma_bf16(const Frag& a, const Frag& b, v8f c) {
  return __builtin_amdgcn_wmma_f32_16x16x32_bf16(false, a.b, false, b.b,
                                                 (short)0, c, false, false);
}

// ---------------------------------------------------------------------------
// Kernel 1: convert the 4 weight matrices fp32 (K x N) -> bf16 transposed (N x K)
// ---------------------------------------------------------------------------
__global__ __launch_bounds__(256)
void convert_weights(const float* __restrict__ WQ, const float* __restrict__ WK,
                     const float* __restrict__ WV, const float* __restrict__ WO,
                     __bf16* __restrict__ WT) {
  int i = blockIdx.x * blockDim.x + threadIdx.x;     // 0 .. 4*65536-1
  if (i >= 4 * 65536) return;
  int w = i >> 16;
  int r = i & 65535;
  int k = r >> 8;
  int n = r & 255;
  const float* src = (w == 0) ? WQ : (w == 1) ? WK : (w == 2) ? WV : WO;
  WT[(size_t)w * 65536 + (size_t)n * 256 + k] = (__bf16)src[(size_t)k * 256 + n];
}

// ---------------------------------------------------------------------------
// Kernel 2: projection GEMM  Y = X(65536x256,f32) @ W(256x256) + bias
// Output scattered bf16 into head-split layout.
//   mode 0: qh/kh  [b][h][s][n][d]
//   mode 1: vhT    [b][h][s][d][n]
// Block tile 128(M) x 128(N), 8 waves: wave = 2 M-tiles x 4 N-tiles.
// ---------------------------------------------------------------------------
__global__ __launch_bounds__(256)
void proj_gemm(const float* __restrict__ A, const __bf16* __restrict__ WT,
               const float* __restrict__ bias, __bf16* __restrict__ Out,
               int mode) {
  const int lane = threadIdx.x & 31;
  const int wv   = threadIdx.x >> 5;   // 0..7
  const int wm   = wv & 3;             // 4 waves along M
  const int wn   = wv >> 2;            // 2 waves along N
  const int l16  = lane & 15;
  const int hi   = lane >> 4;

  const int rowBase = blockIdx.x * 128 + wm * 32;
  const int colBase = blockIdx.y * 128 + wn * 64;

  // Hoisted per-lane pointers (K-step offset folds into instruction offsets)
  const float* ap[2];
#pragma unroll
  for (int t = 0; t < 2; ++t)
    ap[t] = A + (size_t)(rowBase + t * 16 + l16) * 256 + hi * 8;
  const __bf16* bp[4];
#pragma unroll
  for (int nt = 0; nt < 4; ++nt)
    bp[nt] = WT + (size_t)(colBase + nt * 16 + l16) * 256 + hi * 8;

  v8f acc[2][4];
#pragma unroll
  for (int t = 0; t < 2; ++t)
#pragma unroll
    for (int nt = 0; nt < 4; ++nt) acc[t][nt] = vzero8();

#pragma unroll
  for (int k0 = 0; k0 < 256; k0 += 32) {
    // ---- issue ALL loads for this K-step first ----
    Frag bfr[4];
#pragma unroll
    for (int nt = 0; nt < 4; ++nt) {
      bfr[nt].q[0] = *(const v4u*)(bp[nt] + k0);       // B chunks @ hi*8, +16
      bfr[nt].q[1] = *(const v4u*)(bp[nt] + k0 + 16);
    }
    v4f x[2][4];
#pragma unroll
    for (int t = 0; t < 2; ++t) {
      x[t][0] = *(const v4f*)(ap[t] + k0);             // A chunks @ hi*8, +16
      x[t][1] = *(const v4f*)(ap[t] + k0 + 4);
      x[t][2] = *(const v4f*)(ap[t] + k0 + 16);
      x[t][3] = *(const v4f*)(ap[t] + k0 + 20);
      if (k0 < 224) __builtin_prefetch(ap[t] + k0 + 32, 0, 3);  // next K step
    }
    Frag afr[2];
#pragma unroll
    for (int t = 0; t < 2; ++t)
#pragma unroll
      for (int i = 0; i < 4; ++i)
        afr[t].b4[i] = __builtin_convertvector(x[t][i], v4bf);
    // ---- 8 back-to-back WMMAs ----
#pragma unroll
    for (int nt = 0; nt < 4; ++nt) {
      acc[0][nt] = wmma_bf16(afr[0], bfr[nt], acc[0][nt]);
      acc[1][nt] = wmma_bf16(afr[1], bfr[nt], acc[1][nt]);
    }
  }

#pragma unroll
  for (int nt = 0; nt < 4; ++nt) {
    const int col = colBase + nt * 16 + l16;
    const float bv = bias[col];
    const int h = col >> 5, d = col & 31;
#pragma unroll
    for (int t = 0; t < 2; ++t) {
#pragma unroll
      for (int j = 0; j < 8; ++j) {
        const int row = rowBase + t * 16 + j + hi * 8;
        const float val = acc[t][nt][j] + bv;
        const int bb = row >> 13;          // / (S*N)
        const int ss = (row >> 6) & 127;
        const int nn = row & 63;
        size_t addr;
        if (mode == 0)
          addr = ((((size_t)(bb * NHEAD + h) * NSEQ + ss) * NAGENT + nn) * DHEAD + d);
        else
          addr = ((((size_t)(bb * NHEAD + h) * NSEQ + ss) * DHEAD + d) * NAGENT + nn);
        Out[addr] = (__bf16)val;
      }
    }
  }
}

// ---------------------------------------------------------------------------
// Kernel 3: fused attention per (b,h,s): scores = Q K^T / sqrt(32),
// softmax rows -> weights (fp32 to d_out tuple slot + bf16 to LDS),
// attn = W V written bf16 in place over qh.
// 4 waves: wave w owns score rows [16w,16w+16).
// ---------------------------------------------------------------------------
__global__ __launch_bounds__(128)
void attention(__bf16* qh, const __bf16* __restrict__ kh,
               const __bf16* __restrict__ vt, float* __restrict__ wout) {
  __shared__ __bf16 Wlds[64 * 72];   // padded stride 72 (144B, 16B-aligned rows)

  const int bid = blockIdx.x;                // b*H*S + h*S + s
  const size_t base  = (size_t)bid;
  const size_t off2k = base * 2048;          // N*DH = DH*N = 2048 elements
  const int lane = threadIdx.x & 31;
  const int wv   = threadIdx.x >> 5;         // 0..3
  const int l16  = lane & 15;
  const int hi   = lane >> 4;

  // ---- GEMM1: scores = Q @ K^T (M=64, N=64, K=32; single K step) ----
  // Issue all loads first, then 4 back-to-back WMMAs.
  Frag aq, bk[4];
  {
    const __bf16* qp = qh + off2k + (size_t)(wv * 16 + l16) * DHEAD + hi * 8;
    aq.q[0] = *(const v4u*)(qp);
    aq.q[1] = *(const v4u*)(qp + 16);
  }
#pragma unroll
  for (int nt = 0; nt < 4; ++nt) {
    const __bf16* kp = kh + off2k + (size_t)(nt * 16 + l16) * DHEAD + hi * 8;
    bk[nt].q[0] = *(const v4u*)(kp);
    bk[nt].q[1] = *(const v4u*)(kp + 16);
  }
  v8f sc[4];
#pragma unroll
  for (int nt = 0; nt < 4; ++nt)
    sc[nt] = wmma_bf16(aq, bk[nt], vzero8());

  // ---- softmax over columns (m), per row n = wv*16 + j + 8*hi ----
  const float scale = 0.17677669529663687f;  // 1/sqrt(32)
  float wgt[4][8];
#pragma unroll
  for (int j = 0; j < 8; ++j) {
    float mx = -1e30f;
#pragma unroll
    for (int nt = 0; nt < 4; ++nt) {
      float v = sc[nt][j] * scale;
      wgt[nt][j] = v;
      mx = fmaxf(mx, v);
    }
#pragma unroll
    for (int off = 8; off >= 1; off >>= 1) mx = fmaxf(mx, __shfl_xor(mx, off, 32));
    float sum = 0.f;
#pragma unroll
    for (int nt = 0; nt < 4; ++nt) {
      float e = __expf(wgt[nt][j] - mx);
      wgt[nt][j] = e;
      sum += e;
    }
#pragma unroll
    for (int off = 8; off >= 1; off >>= 1) sum += __shfl_xor(sum, off, 32);
    const float inv = 1.0f / sum;
#pragma unroll
    for (int nt = 0; nt < 4; ++nt) wgt[nt][j] *= inv;
  }

  // ---- store weights (fp32 tuple output) + bf16 to LDS ----
  const size_t wbase = base * 4096;          // N*N
#pragma unroll
  for (int nt = 0; nt < 4; ++nt) {
    const int m = nt * 16 + l16;
#pragma unroll
    for (int j = 0; j < 8; ++j) {
      const int n = wv * 16 + j + hi * 8;
      wout[wbase + (size_t)n * 64 + m] = wgt[nt][j];
      Wlds[n * 72 + m] = (__bf16)wgt[nt][j];
    }
  }
  __syncthreads();

  // ---- GEMM2: attn = W @ V (M=64, N=32, K=64; two K steps) ----
  // Preload all 4 V fragments + 2 A fragments, then 4 WMMAs.
  Frag aw[2], bv[2][2];
  const __bf16* wp = &Wlds[(wv * 16 + l16) * 72] + hi * 8;
#pragma unroll
  for (int ks = 0; ks < 2; ++ks) {
    aw[ks].q[0] = *(const v4u*)(wp + ks * 32);
    aw[ks].q[1] = *(const v4u*)(wp + ks * 32 + 16);
  }
#pragma unroll
  for (int dt = 0; dt < 2; ++dt) {
    const __bf16* vp = vt + off2k + (size_t)(dt * 16 + l16) * NAGENT + hi * 8;
#pragma unroll
    for (int ks = 0; ks < 2; ++ks) {
      bv[ks][dt].q[0] = *(const v4u*)(vp + ks * 32);
      bv[ks][dt].q[1] = *(const v4u*)(vp + ks * 32 + 16);
    }
  }
  v8f ac[2] = {vzero8(), vzero8()};
#pragma unroll
  for (int ks = 0; ks < 2; ++ks)
#pragma unroll
    for (int dt = 0; dt < 2; ++dt)
      ac[dt] = wmma_bf16(aw[ks], bv[ks][dt], ac[dt]);

  // ---- write attn bf16 in place over qh[b][h][s][n][d] ----
#pragma unroll
  for (int dt = 0; dt < 2; ++dt) {
    const int d = dt * 16 + l16;
#pragma unroll
    for (int j = 0; j < 8; ++j) {
      const int n = wv * 16 + j + 8 * hi;
      qh[off2k + (size_t)n * DHEAD + d] = (__bf16)ac[dt][j];
    }
  }
}

// ---------------------------------------------------------------------------
// Kernel 4: out = attn(concat heads) @ WO + bias, fp32 output.
// A gathered from head-split bf16 layout [b][h][s][n][d] (k -> h=k>>5, d=k&31).
// ---------------------------------------------------------------------------
__global__ __launch_bounds__(256)
void out_gemm(const __bf16* __restrict__ Abf,
              const __bf16* __restrict__ WT,
              const float* __restrict__ bias, float* __restrict__ Out) {
  const int lane = threadIdx.x & 31;
  const int wv   = threadIdx.x >> 5;
  const int wm   = wv & 3;
  const int wn   = wv >> 2;
  const int l16  = lane & 15;
  const int hi   = lane >> 4;

  const int rowBase = blockIdx.x * 128 + wm * 32;
  const int colBase = blockIdx.y * 128 + wn * 64;

  // Per-lane base for the head-split A gather: the per-head base address for
  // row r; within a head, chunks are at d0 = k & 31 (never crossing heads).
  // addr(h) = (((bb*H + h)*S + ss)*N + nn)*32 + d0 ; hoist h=0 base, step per h
  // is S*N*32 = 262144 elements.
  const __bf16* arow[2];
#pragma unroll
  for (int t = 0; t < 2; ++t) {
    const int row = rowBase + t * 16 + l16;
    const int bb = row >> 13, ss = (row >> 6) & 127, nn = row & 63;
    arow[t] = Abf + ((((size_t)(bb * NHEAD) * NSEQ + ss) * NAGENT + nn) * DHEAD);
  }
  const __bf16* bp[4];
#pragma unroll
  for (int nt = 0; nt < 4; ++nt)
    bp[nt] = WT + (size_t)(colBase + nt * 16 + l16) * 256 + hi * 8;

  v8f acc[2][4];
#pragma unroll
  for (int t = 0; t < 2; ++t)
#pragma unroll
    for (int nt = 0; nt < 4; ++nt) acc[t][nt] = vzero8();

#pragma unroll
  for (int k0 = 0; k0 < 256; k0 += 32) {
    const int kc0 = k0 + hi * 8;             // A chunks
    const int kc1 = k0 + 16 + hi * 8;
    Frag bfr[4];
#pragma unroll
    for (int nt = 0; nt < 4; ++nt) {
      bfr[nt].q[0] = *(const v4u*)(bp[nt] + k0);
      bfr[nt].q[1] = *(const v4u*)(bp[nt] + k0 + 16);
    }
    Frag afr[2];
#pragma unroll
    for (int t = 0; t < 2; ++t) {
      // chunk address: head h = kc>>5 contributes h*S*N*DH, offset d0 = kc&31
      afr[t].q[0] = *(const v4u*)(arow[t] + (size_t)(kc0 >> 5) * (NSEQ*NAGENT*DHEAD) + (kc0 & 31));
      afr[t].q[1] = *(const v4u*)(arow[t] + (size_t)(kc1 >> 5) * (NSEQ*NAGENT*DHEAD) + (kc1 & 31));
    }
#pragma unroll
    for (int nt = 0; nt < 4; ++nt) {
      acc[0][nt] = wmma_bf16(afr[0], bfr[nt], acc[0][nt]);
      acc[1][nt] = wmma_bf16(afr[1], bfr[nt], acc[1][nt]);
    }
  }

#pragma unroll
  for (int nt = 0; nt < 4; ++nt) {
    const int col = colBase + nt * 16 + l16;
    const float bv = bias[col];
#pragma unroll
    for (int t = 0; t < 2; ++t) {
#pragma unroll
      for (int j = 0; j < 8; ++j) {
        const int row = rowBase + t * 16 + j + hi * 8;
        Out[(size_t)row * 256 + col] = acc[t][nt][j] + bv;
      }
    }
  }
}

// ---------------------------------------------------------------------------
extern "C" void kernel_launch(void* const* d_in, const int* in_sizes, int n_in,
                              void* d_out, int out_size, void* d_ws, size_t ws_size,
                              hipStream_t stream) {
  const float* q   = (const float*)d_in[0];
  const float* k   = (const float*)d_in[1];
  const float* v   = (const float*)d_in[2];
  const float* WQw = (const float*)d_in[3];
  const float* WQb = (const float*)d_in[4];
  const float* WKw = (const float*)d_in[5];
  const float* WKb = (const float*)d_in[6];
  const float* WVw = (const float*)d_in[7];
  const float* WVb = (const float*)d_in[8];
  const float* WOw = (const float*)d_in[9];
  const float* WOb = (const float*)d_in[10];

  float* out  = (float*)d_out;
  float* wout = out + OUT_ELEMS;   // tuple element 2: softmax weights

  char* ws = (char*)d_ws;
  __bf16* WT = (__bf16*)ws;                               // 512 KB
  __bf16* qh = (__bf16*)(ws + 524288);                    // 32 MB
  __bf16* kh = (__bf16*)(ws + 524288 + 33554432);         // 32 MB
  __bf16* vt = (__bf16*)(ws + 524288 + 67108864);         // 32 MB

  convert_weights<<<1024, 256, 0, stream>>>(WQw, WKw, WVw, WOw, WT);

  dim3 g(MTOT / 128, DKC / 128);   // 512 x 2
  proj_gemm<<<g, 256, 0, stream>>>(q, WT,           WQb, qh, 0);
  proj_gemm<<<g, 256, 0, stream>>>(k, WT + 65536,   WKb, kh, 0);
  proj_gemm<<<g, 256, 0, stream>>>(v, WT + 131072,  WVb, vt, 1);

  attention<<<NBATCH * NHEAD * NSEQ, 128, 0, stream>>>(qh, kh, vt, wout);

  out_gemm<<<g, 256, 0, stream>>>(qh, WT + 196608, WOb, out);
}
